// EvidenceRefinementModule_919123001747
// MI455X (gfx1250) — matrix operand order repaired and checked by reading, not compiled, not run on hardware
//
#include <hip/hip_runtime.h>

#define DI __device__ __forceinline__

typedef __attribute__((ext_vector_type(16))) _Float16 v16h;
typedef __attribute__((ext_vector_type(8)))  _Float16 v8h;
typedef __attribute__((ext_vector_type(8)))  float    v8f;

// Problem constants
static constexpr int Dm  = 64;     // feature dim
static constexpr int Tit = 3;      // refinement iterations
#define ATT_SCALE 0.25f            // HD^-0.5, HD = 16

// Block config: 256 threads = 8 wave32, 16 batch rows per wave.
static constexpr int WAVES          = 8;
static constexpr int ROWS_PER_BLOCK = WAVES * 16;

// ---- LDS layout -------------------------------------------------------------
// Weights (f16, original [out][in] row-major so B-fragments are contiguous):
static constexpr int W_INIT_OFF = 0;                    // 64 x 128
static constexpr int W_Q_OFF    = W_INIT_OFF + 64*128;  // 64 x 64
static constexpr int W_K_OFF    = W_Q_OFF    + 64*64;
static constexpr int W_V_OFF    = W_K_OFF    + 64*64;
static constexpr int W_O_OFF    = W_V_OFF    + 64*64;
static constexpr int W_IH_OFF   = W_O_OFF    + 64*64;   // 192 x 64
static constexpr int W_HH_OFF   = W_IH_OFF   + 192*64;  // 192 x 64
static constexpr int W_HALFS    = W_HH_OFF   + 192*64;  // 49152 halfs = 96 KB
static constexpr int FLDS_BYTES = 4096;                 // 1024 f32 bias/vec slots
static constexpr int PW_BYTES   = 18944;                // per-wave staging stride
static constexpr int SMEM_BYTES = W_HALFS*2 + FLDS_BYTES + WAVES*PW_BYTES;

DI float sigm(float x) { return 1.0f / (1.0f + __expf(-x)); }

// A fragment: 16x32 f16 tile from row-major LDS (pitch in halfs).
DI v16h load_A(const _Float16* buf, int pitch, int kbase, int lane) {
    const int m = lane & 15, hi = lane >> 4;
    const _Float16* p = buf + m * pitch + kbase + 8 * hi;
    v16h a;
    ((v8h*)&a)[0] = *(const v8h*)(p);       // K = kbase + 8*hi      .. +8
    ((v8h*)&a)[1] = *(const v8h*)(p + 16);  // K = kbase + 16 + 8*hi .. +8
    return a;
}

// B fragment: 32x16 tile; W stored [out][in] row-major, output cols n0..n0+15.
DI v16h load_B(const _Float16* w, int pitch, int n0, int kbase, int lane) {
    const int n = lane & 15, hi = lane >> 4;
    const _Float16* p = w + (n0 + n) * pitch + kbase + 8 * hi;
    v16h b;
    ((v8h*)&b)[0] = *(const v8h*)(p);
    ((v8h*)&b)[1] = *(const v8h*)(p + 16);
    return b;
}

template <int K>
DI v8f gemm(const _Float16* A, int ap, const _Float16* W, int wp, int n0,
            v8f c, int lane) {
#pragma unroll
    for (int kb = 0; kb < K; kb += 32) {
        v16h a = load_A(A, ap, kb, lane);
        v16h b = load_B(W, wp, n0, kb, lane);
        c = __builtin_amdgcn_wmma_f32_16x16x32_f16(false, a, false, b,
                                                   (short)0, c, false, false);
    }
    return c;
}

DI v8f biasC(const float* b, int col0, int lane) {
    const float v = b[col0 + (lane & 15)];
    v8f c;
#pragma unroll
    for (int g = 0; g < 8; ++g) c[g] = v;
    return c;
}

// D tile -> row-major f16 LDS. Element g of lane(hi,n) is (m = g+8*hi, n).
DI void storeD16(_Float16* buf, int pitch, int col0, v8f c, int lane) {
    const int n = lane & 15, hi = lane >> 4;
#pragma unroll
    for (int g = 0; g < 8; ++g)
        buf[(g + 8 * hi) * pitch + col0 + n] = (_Float16)c[g];
}

DI void cvt16(const float* s, _Float16* d, int n, int tid) {
    for (int i = tid; i < n; i += 256) d[i] = (_Float16)s[i];
}
DI void cpyf(const float* s, float* d, int n, int tid) {
    for (int i = tid; i < n; i += 256) d[i] = s[i];
}

__global__ __launch_bounds__(256, 1)
void erm_kernel(const float* __restrict__ badm, const float* __restrict__ aadm,
                const float* __restrict__ W_init, const float* __restrict__ b_init,
                const float* __restrict__ Wq, const float* __restrict__ bq,
                const float* __restrict__ Wk, const float* __restrict__ bk,
                const float* __restrict__ Wv, const float* __restrict__ bv,
                const float* __restrict__ Wo, const float* __restrict__ bo,
                const float* __restrict__ W_ih, const float* __restrict__ W_hh,
                const float* __restrict__ b_ih, const float* __restrict__ b_hh,
                const float* __restrict__ W_pred, const float* __restrict__ b_pred,
                const float* __restrict__ W_fb, const float* __restrict__ b_fb,
                float* __restrict__ out, int B) {
    extern __shared__ __align__(16) char smem[];
    _Float16* wlds = (_Float16*)smem;
    float*    flds = (float*)(smem + W_HALFS * 2);

    const int tid  = threadIdx.x;
    const int lane = tid & 31;
    const int wv   = tid >> 5;

    // ---- one-time per-block: weights -> f16 LDS, biases -> f32 LDS ----
    cvt16(W_init, wlds + W_INIT_OFF, 64 * 128, tid);
    cvt16(Wq,     wlds + W_Q_OFF,    64 * 64,  tid);
    cvt16(Wk,     wlds + W_K_OFF,    64 * 64,  tid);
    cvt16(Wv,     wlds + W_V_OFF,    64 * 64,  tid);
    cvt16(Wo,     wlds + W_O_OFF,    64 * 64,  tid);
    cvt16(W_ih,   wlds + W_IH_OFF,   192 * 64, tid);
    cvt16(W_hh,   wlds + W_HH_OFF,   192 * 64, tid);
    float* B_INIT = flds;        cpyf(b_init, B_INIT, 64,  tid);
    float* BQ  = flds + 64;      cpyf(bq,     BQ,  64,  tid);
    float* BK  = flds + 128;     cpyf(bk,     BK,  64,  tid);
    float* BV  = flds + 192;     cpyf(bv,     BV,  64,  tid);
    float* BO  = flds + 256;     cpyf(bo,     BO,  64,  tid);
    float* BIH = flds + 320;     cpyf(b_ih,   BIH, 192, tid);
    float* BHH = flds + 512;     cpyf(b_hh,   BHH, 192, tid);
    float* WPRED = flds + 704;   cpyf(W_pred, WPRED, 64, tid);
    float* WFB = flds + 768;     cpyf(W_fb,   WFB, 64,  tid);
    float* BFB = flds + 832;     cpyf(b_fb,   BFB, 64,  tid);
    if (tid == 0) { flds[896] = b_pred[0]; flds[1023] = 0.f; }
    __syncthreads();

    // ---- per-wave staging ----
    char* pw = smem + W_HALFS * 2 + FLDS_BYTES + wv * PW_BYTES;
    _Float16* xbuf = (_Float16*)(pw);          // 16 x 128 (evidence, then ctx)
    _Float16* hA   = (_Float16*)(pw + 4096);   // 16 x 64
    _Float16* hB   = (_Float16*)(pw + 6144);   // 16 x 64
    _Float16* kbuf = (_Float16*)(pw + 8192);   // 2 x 16 x 64
    _Float16* vbuf = (_Float16*)(pw + 12288);  // 2 x 16 x 64
    _Float16* qbuf = (_Float16*)(pw + 16384);  // 16 x 64 (Q, then attn ctx)
    float*    pb   = (float*)(pw + 18432);     // 16 x 4 softmax p0 per head
    float*    lb   = (float*)(pw + 18688);     // 16 prev logit
    float*    pp   = (float*)(pw + 18752);     // 16 prev prob

    const int base = blockIdx.x * ROWS_PER_BLOCK + wv * 16;

    // evidence tile -> f16 LDS: [badm | aadm] per row, pitch 128
    for (int i = lane; i < 16 * 64; i += 32) {
        const int r = i >> 6, c = i & 63;
        xbuf[r * 128 + c]      = (_Float16)badm[(size_t)(base + r) * Dm + c];
        xbuf[r * 128 + 64 + c] = (_Float16)aadm[(size_t)(base + r) * Dm + c];
    }
    __syncthreads();

    // h0 = relu([badm|aadm] @ W_init.T + b_init)
#pragma unroll
    for (int ct = 0; ct < 4; ++ct) {
        v8f c = biasC(B_INIT, ct * 16, lane);
        c = gemm<128>(xbuf, 128, wlds + W_INIT_OFF, 128, ct * 16, c, lane);
#pragma unroll
        for (int g = 0; g < 8; ++g) c[g] = fmaxf(c[g], 0.f);
        storeD16(hA, 64, ct * 16, c, lane);
    }
    // K / V for both evidence positions
#pragma unroll
    for (int s = 0; s < 2; ++s) {
#pragma unroll
        for (int ct = 0; ct < 4; ++ct) {
            v8f ck = biasC(BK, ct * 16, lane);
            ck = gemm<64>(xbuf + s * 64, 128, wlds + W_K_OFF, 64, ct * 16, ck, lane);
            storeD16(kbuf + s * 1024, 64, ct * 16, ck, lane);
            v8f cv = biasC(BV, ct * 16, lane);
            cv = gemm<64>(xbuf + s * 64, 128, wlds + W_V_OFF, 64, ct * 16, cv, lane);
            storeD16(vbuf + s * 1024, 64, ct * 16, cv, lane);
        }
    }
    __syncthreads();

    _Float16* hcur = hA;
    _Float16* hnxt = hB;
    for (int t = 0; t < Tit; ++t) {
        // Q = h @ Wq.T + bq
#pragma unroll
        for (int ct = 0; ct < 4; ++ct) {
            v8f c = biasC(BQ, ct * 16, lane);
            c = gemm<64>(hcur, 64, wlds + W_Q_OFF, 64, ct * 16, c, lane);
            storeD16(qbuf, 64, ct * 16, c, lane);
        }
        __syncthreads();

        // attention: softmax over 2 positions; 2 heads per lane
        {
            const int r = lane & 15, hp = (lane >> 4) * 2;
#pragma unroll
            for (int hh = 0; hh < 2; ++hh) {
                const int hd = (hp + hh) * 16;
                float s0 = 0.f, s1 = 0.f;
#pragma unroll
                for (int d = 0; d < 16; ++d) {
                    const float q = (float)qbuf[r * 64 + hd + d];
                    s0 += q * (float)kbuf[r * 64 + hd + d];
                    s1 += q * (float)kbuf[1024 + r * 64 + hd + d];
                }
                s0 *= ATT_SCALE; s1 *= ATT_SCALE;
                const float mx = fmaxf(s0, s1);
                const float e0 = __expf(s0 - mx), e1 = __expf(s1 - mx);
                const float p0 = e0 / (e0 + e1), p1 = 1.f - p0;
                pb[r * 4 + hp + hh] = p0;
#pragma unroll
                for (int d = 0; d < 16; ++d) {
                    const float v0 = (float)vbuf[r * 64 + hd + d];
                    const float v1 = (float)vbuf[1024 + r * 64 + hd + d];
                    qbuf[r * 64 + hd + d] = (_Float16)(p0 * v0 + p1 * v1);
                }
            }
        }
        __syncthreads();

        // attention history (mean over heads; p1-mean = 1 - p0-mean)
        if (lane < 16) {
            const int b = base + lane;
            const float m0 = 0.25f * (pb[lane * 4] + pb[lane * 4 + 1] +
                                      pb[lane * 4 + 2] + pb[lane * 4 + 3]);
            out[(size_t)7 * B + (size_t)b * 6 + t * 2 + 0] = m0;
            out[(size_t)7 * B + (size_t)b * 6 + t * 2 + 1] = 1.f - m0;
        }

        // context = attn_out @ Wo.T + bo (+ logit feedback for t>0) -> xbuf
#pragma unroll
        for (int ct = 0; ct < 4; ++ct) {
            v8f c = biasC(BO, ct * 16, lane);
            c = gemm<64>(qbuf, 64, wlds + W_O_OFF, 64, ct * 16, c, lane);
            if (t > 0) {
                const int n = lane & 15, hi = lane >> 4;
                const float wf = WFB[ct * 16 + n], bf = BFB[ct * 16 + n];
#pragma unroll
                for (int g = 0; g < 8; ++g) c[g] += lb[g + 8 * hi] * wf + bf;
            }
            storeD16(xbuf, 128, ct * 16, c, lane);
        }
        __syncthreads();

        // GRU cell: six gate tiles per 16-column slab, gated in registers
        {
            const int n = lane & 15, hi = lane >> 4;
#pragma unroll
            for (int ct = 0; ct < 4; ++ct) {
                const int c0 = ct * 16;
                v8f gir = gemm<64>(xbuf, 128, wlds + W_IH_OFF, 64, c0,
                                   biasC(BIH, c0, lane), lane);
                v8f ghr = gemm<64>(hcur, 64, wlds + W_HH_OFF, 64, c0,
                                   biasC(BHH, c0, lane), lane);
                v8f giz = gemm<64>(xbuf, 128, wlds + W_IH_OFF, 64, 64 + c0,
                                   biasC(BIH, 64 + c0, lane), lane);
                v8f ghz = gemm<64>(hcur, 64, wlds + W_HH_OFF, 64, 64 + c0,
                                   biasC(BHH, 64 + c0, lane), lane);
                v8f gin = gemm<64>(xbuf, 128, wlds + W_IH_OFF, 64, 128 + c0,
                                   biasC(BIH, 128 + c0, lane), lane);
                v8f ghn = gemm<64>(hcur, 64, wlds + W_HH_OFF, 64, 128 + c0,
                                   biasC(BHH, 128 + c0, lane), lane);
#pragma unroll
                for (int g = 0; g < 8; ++g) {
                    const float r  = sigm(gir[g] + ghr[g]);
                    const float z  = sigm(giz[g] + ghz[g]);
                    const float nn = tanhf(gin[g] + r * ghn[g]);
                    const float ho = (float)hcur[(g + 8 * hi) * 64 + c0 + n];
                    hnxt[(g + 8 * hi) * 64 + c0 + n] =
                        (_Float16)((1.f - z) * nn + z * ho);
                }
            }
        }
        __syncthreads();

        // prediction head + outputs
        if (lane < 16) {
            const int b = base + lane;
            float acc = flds[896];  // b_pred
#pragma unroll
            for (int c = 0; c < 64; ++c)
                acc += (float)hnxt[lane * 64 + c] * WPRED[c];
            const float prob = sigm(acc);
            lb[lane] = acc;
            out[(size_t)B + (size_t)b * 3 + t]     = acc;
            out[(size_t)4 * B + (size_t)b * 3 + t] = prob;
            if (t == Tit - 2) pp[lane] = prob;
            if (t == Tit - 1) {
                out[b] = acc;
                atomicAdd(&flds[1023], fabsf(prob - pp[lane]));
            }
        }
        __syncthreads();

        _Float16* tmp = hcur; hcur = hnxt; hnxt = tmp;
    }

    if (tid == 0)
        atomicAdd(&out[(size_t)13 * B], flds[1023] / (float)B);
}

__global__ void erm_zero_conv(float* __restrict__ out, int B) {
    if (threadIdx.x == 0 && blockIdx.x == 0) out[(size_t)13 * B] = 0.f;
}

extern "C" void kernel_launch(void* const* d_in, const int* in_sizes, int n_in,
                              void* d_out, int out_size, void* d_ws, size_t ws_size,
                              hipStream_t stream) {
    (void)n_in; (void)out_size; (void)d_ws; (void)ws_size;
    const int B = in_sizes[0] / Dm;              // 262144
    const int blocks = B / ROWS_PER_BLOCK;       // 2048

    hipFuncSetAttribute(reinterpret_cast<const void*>(erm_kernel),
                        hipFuncAttributeMaxDynamicSharedMemorySize, SMEM_BYTES);

    erm_zero_conv<<<1, 32, 0, stream>>>((float*)d_out, B);
    erm_kernel<<<blocks, 256, SMEM_BYTES, stream>>>(
        (const float*)d_in[0],  (const float*)d_in[1],
        (const float*)d_in[2],  (const float*)d_in[3],
        (const float*)d_in[4],  (const float*)d_in[5],
        (const float*)d_in[6],  (const float*)d_in[7],
        (const float*)d_in[8],  (const float*)d_in[9],
        (const float*)d_in[10], (const float*)d_in[11],
        (const float*)d_in[12], (const float*)d_in[13],
        (const float*)d_in[14], (const float*)d_in[15],
        (const float*)d_in[16], (const float*)d_in[17],
        (const float*)d_in[18], (const float*)d_in[19],
        (float*)d_out, B);
}
